// ASDNet_51625506898314
// MI455X (gfx1250) — matrix operand order repaired and checked by reading, not compiled, not run on hardware
//
#include <hip/hip_runtime.h>
#include <hip/hip_bf16.h>

typedef __attribute__((ext_vector_type(16))) _Float16     v16h;
typedef __attribute__((ext_vector_type(8)))  _Float16     v8h;
typedef __attribute__((ext_vector_type(8)))  float        v8f;
typedef __attribute__((ext_vector_type(4)))  unsigned int v4u;
typedef __attribute__((ext_vector_type(8)))  unsigned int v8u;

#define EPSBN 1e-5f

// ---------------------------------------------------------------------------
// Involution block (3 ch, k=3, hidden=1).  Tiny FLOPs -> scalar VALU.
// ---------------------------------------------------------------------------
__global__ void involution_k(const float* __restrict__ X, float* __restrict__ Y,
                             const float* __restrict__ w1, const float* __restrict__ b1,
                             const float* __restrict__ g,  const float* __restrict__ be,
                             const float* __restrict__ m,  const float* __restrict__ v,
                             const float* __restrict__ w2, const float* __restrict__ b2,
                             int Bimg, int lyr) {
    const int H = 46, W = 46;
    int idx = blockIdx.x * blockDim.x + threadIdx.x;
    int tot = Bimg * H * W;
    if (idx >= tot) return;
    int ww = idx % W;
    int hh = (idx / W) % H;
    int b  = idx / (W * H);
    const float* xb = X + (long long)b * 3 * H * W;
    int p = hh * W + ww;
    float x0 = xb[0 * H * W + p];
    float x1 = xb[1 * H * W + p];
    float x2 = xb[2 * H * W + p];
    float t = x0 * w1[lyr * 3 + 0] + x1 * w1[lyr * 3 + 1] + x2 * w1[lyr * 3 + 2] + b1[lyr];
    float sc = g[lyr] * rsqrtf(v[lyr] + EPSBN);
    t = (t - m[lyr]) * sc + be[lyr];
    t = t > 0.f ? t : 0.f;
    float kern[9];
#pragma unroll
    for (int k = 0; k < 9; ++k) kern[k] = w2[lyr * 9 + k] * t + b2[lyr * 9 + k];
#pragma unroll
    for (int c = 0; c < 3; ++c) {
        const float* xc = xb + c * H * W;
        float s = 0.f;
#pragma unroll
        for (int i = 0; i < 3; ++i) {
#pragma unroll
            for (int j = 0; j < 3; ++j) {
                int hi = hh + i - 1, wi = ww + j - 1;
                float xv = (hi >= 0 && hi < H && wi >= 0 && wi < W) ? xc[hi * W + wi] : 0.f;
                s += kern[i * 3 + j] * xv;
            }
        }
        Y[((long long)b * 3 + c) * H * W + p] = s;
    }
}

// ---------------------------------------------------------------------------
__global__ void cvt_f32_f16_k(const float* __restrict__ X, _Float16* __restrict__ Y,
                              long long n, int doRelu) {
    long long i = (long long)blockIdx.x * blockDim.x + threadIdx.x;
    if (i >= n) return;
    float val = X[i];
    if (doRelu) val = val > 0.f ? val : 0.f;
    Y[i] = (_Float16)val;
}

// ---------------------------------------------------------------------------
__global__ void pack_w_k(const float* __restrict__ src, _Float16* __restrict__ dst,
                         int Cout, int Kreal, int Kpad) {
    int i = blockIdx.x * blockDim.x + threadIdx.x;
    if (i >= Cout * Kpad) return;
    int co = i / Kpad, k = i - co * Kpad;
    dst[i] = (k < Kreal) ? (_Float16)src[co * Kreal + k] : (_Float16)0.f;
}

// ---------------------------------------------------------------------------
__global__ void bn_fold_k(const float* __restrict__ cb, const float* __restrict__ g,
                          const float* __restrict__ be, const float* __restrict__ m,
                          const float* __restrict__ v, float* __restrict__ alpha,
                          float* __restrict__ beta, int C) {
    int c = blockIdx.x * blockDim.x + threadIdx.x;
    if (c >= C) return;
    float a = g[c] * rsqrtf(v[c] + EPSBN);
    alpha[c] = a;
    beta[c]  = (cb[c] - m[c]) * a + be[c];
}

// ---------------------------------------------------------------------------
// Implicit-GEMM 3x3 VALID conv via V_WMMA_F32_16X16X32_F16.
//  * All 4 waves of a block share one 16-row M-tile; its weights (16 x Kpad
//    f16, contiguous in memory) are staged into LDS ONCE by the Tensor Data
//    Mover (tensor_load_to_lds, waited with s_wait_tensorcnt), so A-fragment
//    reads are ds_load_b128 and the VMEM pipe serves only the B gather.
//  * im2col K->offset table in LDS; branchless clamped gather; padded-K slots
//    point at offset 0 and are annihilated by zero-padded weight columns.
//  * EXEC is all-ones at every wmma.
// ---------------------------------------------------------------------------
__global__ __launch_bounds__(128) void conv3x3_wmma_k(
    const _Float16* __restrict__ X, const _Float16* __restrict__ Wh,
    const float* __restrict__ alpha, const float* __restrict__ beta,
    _Float16* __restrict__ Y,
    int Bimg, int Cin, int IH, int IW, int OH, int OW, int Cout,
    int Kreal, int Kpad) {
    __shared__ int      offs[576];          // Kpad <= 576 for all layers
    __shared__ _Float16 Wlds[16 * 576];     // one 16-row A-tile (<= 18 KB)

    const int tilesM = Cout >> 4;
    const long long Ntot   = (long long)Bimg * OH * OW;
    const long long tilesN = (Ntot + 15) >> 4;
    const int       tm     = blockIdx.x % tilesM;           // shared by all waves
    const long long tnBase = (long long)(blockIdx.x / tilesM) * 4;

    for (int k = threadIdx.x; k < Kpad; k += blockDim.x) {
        int ci = k / 9;
        int rs = k - ci * 9;
        int r  = rs / 3;
        int s  = rs - r * 3;
        offs[k] = (k < Kreal) ? (ci * IH + r) * IW + s : 0;
    }

    // --- TDM: DMA this block's weight tile into LDS (wave 0 only) ---
    if ((threadIdx.x >> 5) == 0) {
        const unsigned int nE = (unsigned int)(16 * Kpad);          // elements (2B each)
        unsigned long long ga =
            (unsigned long long)(uintptr_t)(Wh + (size_t)tm * 16 * Kpad);
        unsigned int ldsa = (unsigned int)(uintptr_t)(&Wlds[0]);    // LDS byte offset
        v4u g0;
        g0[0] = 1u;                                  // count=1, user descriptor
        g0[1] = ldsa;                                // lds_addr
        g0[2] = (unsigned int)(ga & 0xffffffffull);  // global_addr[31:0]
        g0[3] = (unsigned int)((ga >> 32) & 0x01ffffffull) | (2u << 30); // addr[56:32] | type=2
        v8u g1;
        g1[0] = 0x00010000u;                         // wg_mask=0, data_size=1 (2 bytes)
        g1[1] = (nE & 0xffffu) << 16;                // tensor_dim0[15:0]
        g1[2] = ((nE >> 16) & 0xffffu) | (1u << 16); // tensor_dim0[31:16] | tensor_dim1=1
        g1[3] = (nE & 0xffffu) << 16;                // tile_dim0 = nE (fits 16 bits, max 9216)
        g1[4] = 1u;                                  // tile_dim1=1, tile_dim2=0
        g1[5] = nE;                                  // tensor_dim0_stride[31:0]
        g1[6] = 0u;                                  // stride hi / dim1_stride lo
        g1[7] = 0u;
        asm volatile("tensor_load_to_lds %0, %1" :: "s"(g0), "s"(g1) : "memory");
        __builtin_amdgcn_s_wait_tensorcnt(0);
    }
    __syncthreads();

    const int lane = threadIdx.x & 31;
    const int wv   = threadIdx.x >> 5;
    const long long tn = tnBase + wv;
    if (tn >= tilesN) return;                         // wave-uniform

    const int hsel = lane >> 4;   // 0 = lanes 0-15, 1 = lanes 16-31
    const int l16  = lane & 15;

    // --- this lane's output column (pixel); clamp keeps all loads in-bounds
    long long n   = tn * 16 + l16;
    long long nc  = (n < Ntot) ? n : (Ntot - 1);
    const int ohw = OH * OW;
    const int b   = (int)(nc / ohw);
    const int rm  = (int)(nc % ohw);
    const int oh  = rm / OW;
    const int ow  = rm - oh * OW;
    const _Float16* xbase = X + ((long long)b * Cin * IH + oh) * IW + ow;

    // --- A row from LDS: M-row = l16, halves cover K = kb+hsel*8+(0..7)
    //     and K = kb+16+hsel*8+(0..7)  (ISA 16-bit A 16x32 layout)
    const _Float16* wrow = &Wlds[l16 * Kpad + hsel * 8];

    v8f acc = {};
    for (int kb = 0; kb < Kpad; kb += 32) {
        union { v16h v; v8h h8[2]; } A;
        A.h8[0] = *(const v8h*)(wrow + kb);           // ds_load_b128
        A.h8[1] = *(const v8h*)(wrow + kb + 16);

        // B fragment: halves hold K = kb+hsel*8+(0..7) and K = kb+16+hsel*8+(0..7)
        const int kk0 = kb + hsel * 8;                // multiple of 8 -> int4 aligned
        const int4* op = (const int4*)&offs[kk0];
        int4 o0 = op[0], o1 = op[1], o2 = op[4], o3 = op[5];
        union { v16h v; _Float16 h[16]; } Bf;
        Bf.h[0]  = xbase[o0.x]; Bf.h[1]  = xbase[o0.y];
        Bf.h[2]  = xbase[o0.z]; Bf.h[3]  = xbase[o0.w];
        Bf.h[4]  = xbase[o1.x]; Bf.h[5]  = xbase[o1.y];
        Bf.h[6]  = xbase[o1.z]; Bf.h[7]  = xbase[o1.w];
        Bf.h[8]  = xbase[o2.x]; Bf.h[9]  = xbase[o2.y];
        Bf.h[10] = xbase[o2.z]; Bf.h[11] = xbase[o2.w];
        Bf.h[12] = xbase[o3.x]; Bf.h[13] = xbase[o3.y];
        Bf.h[14] = xbase[o3.z]; Bf.h[15] = xbase[o3.w];

        acc = __builtin_amdgcn_wmma_f32_16x16x32_f16(false, A.v, false, Bf.v,
                                                     (short)0, acc, false, false);
    }

    if (n >= Ntot) return;   // divergence only after all wmma issued
    // D layout: VGPR v -> row M = tm*16 + hsel*8 + v, col N = l16
#pragma unroll
    for (int vi = 0; vi < 8; ++vi) {
        int co = tm * 16 + hsel * 8 + vi;
        float y = acc[vi] * alpha[co] + beta[co];
        y = y > 0.f ? y : 0.f;
        Y[(((long long)b * Cout + co) * OH + oh) * OW + ow] = (_Float16)y;
    }
}

// ---------------------------------------------------------------------------
__global__ void maxpool2_k(const _Float16* __restrict__ X, _Float16* __restrict__ Y,
                           int BC, int IH, int IW) {
    int OH = IH >> 1, OW = IW >> 1;
    long long tot = (long long)BC * OH * OW;
    long long i = (long long)blockIdx.x * blockDim.x + threadIdx.x;
    if (i >= tot) return;
    int ow = (int)(i % OW);
    int oh = (int)((i / OW) % OH);
    long long bc = i / ((long long)OW * OH);
    const _Float16* xp = X + (bc * IH + oh * 2) * IW + ow * 2;
    float a = (float)xp[0], b = (float)xp[1], c = (float)xp[IW], d = (float)xp[IW + 1];
    float mx = fmaxf(fmaxf(a, b), fmaxf(c, d));
    Y[i] = (_Float16)mx;
}

// ---------------------------------------------------------------------------
// fc1: D[M=128, N=1024], K=2048 WMMA GEMM; A and B loads fully vectorized.
// ---------------------------------------------------------------------------
__global__ __launch_bounds__(128) void fc1_wmma_k(
    const _Float16* __restrict__ X,   // (N, K) row-major activations
    const _Float16* __restrict__ Wh,  // (M, K) row-major f16 weights
    const float* __restrict__ bias,
    _Float16* __restrict__ Y,         // (N, M)
    int N, int M, int K) {
    const int lane = threadIdx.x & 31;
    const int wv   = threadIdx.x >> 5;
    const int tilesM = M >> 4;
    const int tilesN = N >> 4;
    int tile = blockIdx.x * 4 + wv;
    if (tile >= tilesM * tilesN) return;
    const int tm = tile % tilesM;
    const int tn = tile / tilesM;

    const int hsel = lane >> 4;
    const int l16  = lane & 15;
    const int n    = tn * 16 + l16;

    const _Float16* wrow = Wh + (long long)(tm * 16 + l16) * K + hsel * 8;
    const _Float16* xrow = X + (long long)n * K + hsel * 8;

    v8f acc = {};
    for (int kb = 0; kb < K; kb += 32) {
        union { v16h v; v8h h8[2]; } A;
        A.h8[0] = *(const v8h*)(wrow + kb);
        A.h8[1] = *(const v8h*)(wrow + kb + 16);
        union { v16h v; v8h h8[2]; } Bf;
        Bf.h8[0] = *(const v8h*)(xrow + kb);
        Bf.h8[1] = *(const v8h*)(xrow + kb + 16);
        acc = __builtin_amdgcn_wmma_f32_16x16x32_f16(false, A.v, false, Bf.v,
                                                     (short)0, acc, false, false);
    }
#pragma unroll
    for (int vi = 0; vi < 8; ++vi) {
        int co = tm * 16 + hsel * 8 + vi;
        float y = acc[vi] + bias[co];
        y = y > 0.f ? y : 0.f;
        Y[(long long)n * M + co] = (_Float16)y;
    }
}

// ---------------------------------------------------------------------------
__global__ void fc2_k(const _Float16* __restrict__ A, const float* __restrict__ W,
                      const float* __restrict__ bias, float* __restrict__ out, int N) {
    int i = blockIdx.x * blockDim.x + threadIdx.x;
    if (i >= N * 2) return;
    int b = i >> 1, o = i & 1;
    float s = bias[o];
#pragma unroll 8
    for (int k = 0; k < 128; ++k) s += (float)A[b * 128 + k] * W[o * 128 + k];
    out[i] = s;
}

// ---------------------------------------------------------------------------
extern "C" void kernel_launch(void* const* d_in, const int* in_sizes, int n_in,
                              void* d_out, int out_size, void* d_ws, size_t ws_size,
                              hipStream_t stream) {
    (void)in_sizes; (void)n_in; (void)out_size; (void)ws_size;
    const int B = 1024;

    const float* x      = (const float*)d_in[0];
    const float* inv_w1 = (const float*)d_in[1];
    const float* inv_b1 = (const float*)d_in[2];
    const float* inv_g  = (const float*)d_in[3];
    const float* inv_be = (const float*)d_in[4];
    const float* inv_m  = (const float*)d_in[5];
    const float* inv_v  = (const float*)d_in[6];
    const float* inv_w2 = (const float*)d_in[7];
    const float* inv_b2 = (const float*)d_in[8];
    const float* cw1 = (const float*)d_in[9];  const float* cb1 = (const float*)d_in[10];
    const float* g1  = (const float*)d_in[11]; const float* be1 = (const float*)d_in[12];
    const float* m1  = (const float*)d_in[13]; const float* v1  = (const float*)d_in[14];
    const float* cw2 = (const float*)d_in[15]; const float* cb2 = (const float*)d_in[16];
    const float* g2  = (const float*)d_in[17]; const float* be2 = (const float*)d_in[18];
    const float* m2  = (const float*)d_in[19]; const float* v2  = (const float*)d_in[20];
    const float* cw3 = (const float*)d_in[21]; const float* cb3 = (const float*)d_in[22];
    const float* g3  = (const float*)d_in[23]; const float* be3 = (const float*)d_in[24];
    const float* m3  = (const float*)d_in[25]; const float* v3  = (const float*)d_in[26];
    const float* fw1 = (const float*)d_in[27]; const float* fb1 = (const float*)d_in[28];
    const float* fw2 = (const float*)d_in[29]; const float* fb2 = (const float*)d_in[30];

    char* ws = (char*)d_ws;
    size_t off = 0;
    auto bump = [&](size_t bytes) { size_t o = off; off += (bytes + 255) & ~(size_t)255; return o; };
    const size_t oR0 = bump((size_t)B * 3 * 46 * 46 * 4);   // f32 involution ping / later f16 pool2
    const size_t oR1 = bump((size_t)B * 3 * 46 * 46 * 4);   // f32 involution pong / later f16 conv3-out
    const size_t oR2 = bump((size_t)B * 3 * 46 * 46 * 2 + 65536); // f16 conv1 input (+slack) / pool3
    const size_t oR3 = bump((size_t)B * 32 * 44 * 44 * 2);  // f16 conv1-out / conv2-out
    const size_t oR4 = bump((size_t)B * 32 * 22 * 22 * 2);  // f16 pool1 / fc1-out
    const size_t oW1 = bump((size_t)32 * 32 * 2);
    const size_t oW2 = bump((size_t)64 * 288 * 2);
    const size_t oW3 = bump((size_t)128 * 576 * 2);
    const size_t oFW = bump((size_t)128 * 2048 * 2);
    const size_t oA1 = bump(32 * 4),  oB1 = bump(32 * 4);
    const size_t oA2 = bump(64 * 4),  oB2 = bump(64 * 4);
    const size_t oA3 = bump(128 * 4), oB3 = bump(128 * 4);

    float*    R0f = (float*)(ws + oR0);
    float*    R1f = (float*)(ws + oR1);
    _Float16* R0h = (_Float16*)(ws + oR0);
    _Float16* R1h = (_Float16*)(ws + oR1);
    _Float16* R2h = (_Float16*)(ws + oR2);
    _Float16* R3h = (_Float16*)(ws + oR3);
    _Float16* R4h = (_Float16*)(ws + oR4);
    _Float16* W1h = (_Float16*)(ws + oW1);
    _Float16* W2h = (_Float16*)(ws + oW2);
    _Float16* W3h = (_Float16*)(ws + oW3);
    _Float16* FWh = (_Float16*)(ws + oFW);
    float *A1 = (float*)(ws + oA1), *B1 = (float*)(ws + oB1);
    float *A2 = (float*)(ws + oA2), *B2 = (float*)(ws + oB2);
    float *A3 = (float*)(ws + oA3), *B3 = (float*)(ws + oB3);

    // ---- parameter prep ----
    pack_w_k<<<(32 * 32 + 255) / 256, 256, 0, stream>>>(cw1, W1h, 32, 27, 32);
    pack_w_k<<<(64 * 288 + 255) / 256, 256, 0, stream>>>(cw2, W2h, 64, 288, 288);
    pack_w_k<<<(128 * 576 + 255) / 256, 256, 0, stream>>>(cw3, W3h, 128, 576, 576);
    cvt_f32_f16_k<<<(128 * 2048 + 255) / 256, 256, 0, stream>>>(fw1, FWh, 128 * 2048, 0);
    bn_fold_k<<<1, 64, 0, stream>>>(cb1, g1, be1, m1, v1, A1, B1, 32);
    bn_fold_k<<<1, 64, 0, stream>>>(cb2, g2, be2, m2, v2, A2, B2, 64);
    bn_fold_k<<<2, 64, 0, stream>>>(cb3, g3, be3, m3, v3, A3, B3, 128);

    // ---- 3 involution blocks (ping-pong) + ReLU + f16 convert ----
    const int pixTot = B * 46 * 46;
    involution_k<<<(pixTot + 255) / 256, 256, 0, stream>>>(x,   R0f, inv_w1, inv_b1, inv_g,
                                                           inv_be, inv_m, inv_v, inv_w2, inv_b2, B, 0);
    involution_k<<<(pixTot + 255) / 256, 256, 0, stream>>>(R0f, R1f, inv_w1, inv_b1, inv_g,
                                                           inv_be, inv_m, inv_v, inv_w2, inv_b2, B, 1);
    involution_k<<<(pixTot + 255) / 256, 256, 0, stream>>>(R1f, R0f, inv_w1, inv_b1, inv_g,
                                                           inv_be, inv_m, inv_v, inv_w2, inv_b2, B, 2);
    cvt_f32_f16_k<<<((long long)B * 3 * 2116 + 255) / 256, 256, 0, stream>>>(
        R0f, R2h, (long long)B * 3 * 2116, 1);

    auto conv = [&](const _Float16* Xa, const _Float16* Wm, const float* al, const float* bt,
                    _Float16* Ya, int Cin, int IH, int IW, int Cout, int Kreal, int Kpad) {
        int OH = IH - 2, OW = IW - 2;
        long long Ntot = (long long)B * OH * OW;
        long long tilesM = Cout / 16;
        long long tilesN = (Ntot + 15) / 16;
        int blocks = (int)(tilesM * ((tilesN + 3) / 4));
        conv3x3_wmma_k<<<blocks, 128, 0, stream>>>(Xa, Wm, al, bt, Ya, B, Cin, IH, IW,
                                                   OH, OW, Cout, Kreal, Kpad);
    };

    // conv1 (3->32, 46->44) + BN + ReLU, pool -> 22x22
    conv(R2h, W1h, A1, B1, R3h, 3, 46, 46, 32, 27, 32);
    maxpool2_k<<<((long long)B * 32 * 22 * 22 + 255) / 256, 256, 0, stream>>>(R3h, R4h, B * 32, 44, 44);
    // conv2 (32->64, 22->20) + BN + ReLU, pool -> 10x10
    conv(R4h, W2h, A2, B2, R3h, 32, 22, 22, 64, 288, 288);
    maxpool2_k<<<((long long)B * 64 * 10 * 10 + 255) / 256, 256, 0, stream>>>(R3h, R0h, B * 64, 20, 20);
    // conv3 (64->128, 10->8) + BN + ReLU, pool -> 4x4
    conv(R0h, W3h, A3, B3, R1h, 64, 10, 10, 128, 576, 576);
    maxpool2_k<<<((long long)B * 128 * 4 * 4 + 255) / 256, 256, 0, stream>>>(R1h, R2h, B * 128, 8, 8);

    // fc1 (2048 -> 128) WMMA + ReLU, then fc2 (128 -> 2) scalar
    fc1_wmma_k<<<(8 * 64) / 4, 128, 0, stream>>>(R2h, FWh, fb1, R4h, B, 128, 2048);
    fc2_k<<<(B * 2 + 255) / 256, 256, 0, stream>>>(R4h, fw2, fb2, (float*)d_out, B);
}
